// cust_60455959658671
// MI455X (gfx1250) — matrix operand order repaired and checked by reading, not compiled, not run on hardware
//
#include <hip/hip_runtime.h>
#include <hip/hip_bf16.h>

#define B_    64
#define S_    512
#define E_    256
#define H_    256
#define G4H   1024     // 4*H
#define TWO_H 512
#define L_    5
#define V_    50000

typedef __attribute__((ext_vector_type(16))) __bf16        v16bf;
typedef __attribute__((ext_vector_type(8)))  float         v8f;
typedef __attribute__((ext_vector_type(4)))  unsigned int  v4u;

union Frag {
    v16bf v;
    v4u   q[2];
};

// ---------------------------------------------------------------------------
// f32 -> bf16 conversion (grid-stride)
// ---------------------------------------------------------------------------
__global__ void cvt_f32_bf16(const float* __restrict__ src,
                             __hip_bfloat16* __restrict__ dst, int n) {
    int i = blockIdx.x * blockDim.x + threadIdx.x;
    int stride = gridDim.x * blockDim.x;
    for (; i < n; i += stride) dst[i] = __float2bfloat16(src[i]);
}

__global__ void add_f32(const float* __restrict__ a, const float* __restrict__ b,
                        float* __restrict__ o, int n) {
    int i = blockIdx.x * blockDim.x + threadIdx.x;
    if (i < n) o[i] = a[i] + b[i];
}

// ---------------------------------------------------------------------------
// Input projection: G[s,b,:] = embed[x[b, s']] @ wih^T + (bih+bhh)
// One wave -> one 16x16 D tile.  M = (s,b) flattened (32768), N = 4H (1024).
// ---------------------------------------------------------------------------
__global__ void __launch_bounds__(256)
input_proj(const __hip_bfloat16* __restrict__ embed,   // [V,E] bf16
           const __hip_bfloat16* __restrict__ wih,     // [4H,E] bf16 (row-major N,K)
           const float* __restrict__ bias,             // [4H] = bih+bhh
           const int* __restrict__ x,                  // [B,S]
           const int* __restrict__ lens,               // [B]
           __hip_bfloat16* __restrict__ G,             // [S,B,4H] bf16
           int backward)
{
    int lane = threadIdx.x & 31;
    int wave = threadIdx.x >> 5;
    int half = lane >> 4;
    int lm   = lane & 15;

    int mtile = blockIdx.x >> 3;                 // 0..2047  (16 rows each)
    int ntile = ((blockIdx.x & 7) << 3) + wave;  // 0..63

    int s  = mtile >> 2;                         // 16-row tile => single s
    int b0 = (mtile & 3) * 16;

    // A row for this lane (rows m = lane%16)
    int b  = b0 + lm;
    int sp = s;
    if (backward) {
        int len = lens[b];
        sp = (s < len) ? (len - 1 - s) : s;
    }
    int tok = x[b * S_ + sp];
    const __hip_bfloat16* arow = embed + (size_t)tok * E_;
    const __hip_bfloat16* brow = wih + (size_t)(ntile * 16 + lm) * E_;

    v8f acc = {};
    #pragma unroll
    for (int kb = 0; kb < E_; kb += 32) {
        Frag A, Bf;
        A.q[0]  = *(const v4u*)(arow + kb + half * 8);
        A.q[1]  = *(const v4u*)(arow + kb + 16 + half * 8);
        Bf.q[0] = *(const v4u*)(brow + kb + half * 8);
        Bf.q[1] = *(const v4u*)(brow + kb + 16 + half * 8);
        acc = __builtin_amdgcn_wmma_f32_16x16x32_bf16(
                  false, A.v, false, Bf.v, (short)0, acc, false, false);
    }

    int n = ntile * 16 + lm;
    float bv = bias[n];
    #pragma unroll
    for (int r = 0; r < 8; ++r) {
        int bb = b0 + 8 * half + r;
        G[((size_t)s * B_ + bb) * G4H + n] = __float2bfloat16(acc[r] + bv);
    }
}

// ---------------------------------------------------------------------------
// Persistent recurrent kernel: one workgroup per direction, 1024 threads
// (32 waves).  Per step: gates(LDS f32) = h(LDS bf16) @ whh^T, then the
// element-wise LSTM cell with c kept in registers.
// LDS: 256KB gates + 32KB h = 288KB dynamic (<= 320KB WGP LDS) + 256B static
// Each wave owns 2 N-tiles; per N-tile the 8 K-block B fragments of whh are
// cached in registers and reused across the 4 M-tiles (4x less L2 traffic).
// ---------------------------------------------------------------------------
#define REC_SMEM (B_ * G4H * 4 + B_ * H_ * 2)

__global__ void __launch_bounds__(1024)
lstm_rec(const __hip_bfloat16* __restrict__ whh_f,
         const __hip_bfloat16* __restrict__ whh_b,
         const __hip_bfloat16* __restrict__ Gf,
         const __hip_bfloat16* __restrict__ Gb,
         const int* __restrict__ lens,
         float* __restrict__ Hout)                 // [B,S,2H] f32
{
    extern __shared__ char smem[];
    float*          gates = (float*)smem;                           // [64][1024]
    __hip_bfloat16* hbuf  = (__hip_bfloat16*)(smem + B_ * G4H * 4); // [64][256]
    __shared__ int  lensh[B_];

    const int dir = blockIdx.x;
    const __hip_bfloat16* whh = dir ? whh_b : whh_f;
    const __hip_bfloat16* G   = dir ? Gb : Gf;

    const int tid  = threadIdx.x;
    const int lane = tid & 31;
    const int wave = tid >> 5;
    const int half = lane >> 4;
    const int lm   = lane & 15;

    for (int i = tid; i < B_ * H_; i += 1024) hbuf[i] = __float2bfloat16(0.0f);
    if (tid < B_) lensh[tid] = lens[tid];
    float c[16];
    #pragma unroll
    for (int q = 0; q < 16; ++q) c[q] = 0.0f;
    __syncthreads();

    for (int t = 0; t < S_; ++t) {
        // prefetch next step's G slab (128KB across 1024 threads) into cache
        if (t + 1 < S_) {
            const char* pf = (const char*)(G + (size_t)(t + 1) * (B_ * G4H))
                             + tid * 128;
            __builtin_prefetch(pf, 0, 1);
        }

        // ---- phase 1: gates = h @ whh^T  (M=64, N=1024, K=256) ------------
        #pragma unroll
        for (int nn = 0; nn < 2; ++nn) {
            int ntile = wave * 2 + nn;
            const __hip_bfloat16* brow = whh + (size_t)(ntile * 16 + lm) * H_;
            Frag Bfr[8];
            #pragma unroll
            for (int kb = 0; kb < 8; ++kb) {
                Bfr[kb].q[0] = *(const v4u*)(brow + kb * 32 + half * 8);
                Bfr[kb].q[1] = *(const v4u*)(brow + kb * 32 + 16 + half * 8);
            }
            #pragma unroll
            for (int mtile = 0; mtile < 4; ++mtile) {
                const __hip_bfloat16* arow = hbuf + (mtile * 16 + lm) * H_;
                v8f acc = {};
                #pragma unroll
                for (int kb = 0; kb < 8; ++kb) {
                    Frag A;
                    A.q[0] = *(const v4u*)(arow + kb * 32 + half * 8);
                    A.q[1] = *(const v4u*)(arow + kb * 32 + 16 + half * 8);
                    acc = __builtin_amdgcn_wmma_f32_16x16x32_bf16(
                              false, A.v, false, Bfr[kb].v, (short)0, acc,
                              false, false);
                }
                int n = ntile * 16 + lm;
                #pragma unroll
                for (int r = 0; r < 8; ++r) {
                    int m = mtile * 16 + 8 * half + r;
                    gates[m * G4H + n] = acc[r];
                }
            }
        }
        __syncthreads();

        // ---- phase 2: LSTM cell -------------------------------------------
        const __hip_bfloat16* Gt = G + (size_t)t * (B_ * G4H);
        #pragma unroll
        for (int q = 0; q < 16; ++q) {
            int p  = q * 1024 + tid;      // 16384 (b,j) pairs
            int bb = p >> 8;
            int j  = p & 255;
            const float*          gr = gates + bb * G4H;
            const __hip_bfloat16* gg = Gt + (size_t)bb * G4H;
            float pi = gr[j]       + __bfloat162float(gg[j]);
            float pf = gr[256 + j] + __bfloat162float(gg[256 + j]);
            float pg = gr[512 + j] + __bfloat162float(gg[512 + j]);
            float po = gr[768 + j] + __bfloat162float(gg[768 + j]);
            float ig = 1.0f / (1.0f + __expf(-pi));
            float fg = 1.0f / (1.0f + __expf(-pf));
            float og = 1.0f / (1.0f + __expf(-po));
            float cn = fg * c[q] + ig * tanhf(pg);
            c[q] = cn;
            float hn = og * tanhf(cn);

            int spos = t;
            if (dir) { int len = lensh[bb]; spos = (t < len) ? (len - 1 - t) : t; }
            Hout[((size_t)bb * S_ + spos) * TWO_H + dir * H_ + j] = hn;
            hbuf[bb * H_ + j] = __float2bfloat16(hn);
        }
        __syncthreads();
    }
}

// ---------------------------------------------------------------------------
// Attention + final projection.  One block per batch element.
// ---------------------------------------------------------------------------
__global__ void __launch_bounds__(512)
attn_out(const float* __restrict__ Hout,       // [B,S,2H]
         const float* __restrict__ att_usr,    // [NU,2H]
         const float* __restrict__ att_prd,    // [NP,2H]
         const float* __restrict__ W,          // [L,2H]
         const float* __restrict__ bvec,       // [1,L]
         const int* __restrict__ lens,
         const int* __restrict__ usr,
         const int* __restrict__ prd,
         float* __restrict__ out)              // [B,L]
{
    __shared__ __align__(16) float meta[TWO_H];
    __shared__ float sc[S_];
    __shared__ float red[512];
    __shared__ float ctx[TWO_H];

    const int b   = blockIdx.x;
    const int tid = threadIdx.x;
    const int len = lens[b];

    meta[tid] = att_usr[(size_t)usr[b] * TWO_H + tid] +
                att_prd[(size_t)prd[b] * TWO_H + tid];
    __syncthreads();

    const float* hb = Hout + (size_t)b * S_ * TWO_H;
    const float4* rowv  = (const float4*)(hb + (size_t)tid * TWO_H);
    const float4* metav = (const float4*)meta;
    float acc = 0.0f;
    #pragma unroll 4
    for (int k = 0; k < TWO_H / 4; ++k) {
        float4 hv = rowv[k];
        float4 mv = metav[k];
        acc += hv.x * mv.x + hv.y * mv.y + hv.z * mv.z + hv.w * mv.w;
    }
    if (tid >= len) acc = -1e9f;
    sc[tid]  = acc;
    red[tid] = acc;
    __syncthreads();
    for (int off = 256; off > 0; off >>= 1) {
        if (tid < off) red[tid] = fmaxf(red[tid], red[tid + off]);
        __syncthreads();
    }
    float mx = red[0];
    __syncthreads();
    float e  = __expf(sc[tid] - mx);
    sc[tid]  = e;
    red[tid] = e;
    __syncthreads();
    for (int off = 256; off > 0; off >>= 1) {
        if (tid < off) red[tid] += red[tid + off];
        __syncthreads();
    }
    float inv = 1.0f / red[0];
    sc[tid] *= inv;
    __syncthreads();

    float a2 = 0.0f;
    for (int s = 0; s < S_; ++s) a2 += sc[s] * hb[(size_t)s * TWO_H + tid];
    ctx[tid] = a2;
    __syncthreads();

    if (tid < L_) {
        float o = bvec[tid];
        for (int k = 0; k < TWO_H; ++k) o += ctx[k] * W[tid * TWO_H + k];
        out[b * L_ + tid] = o;
    }
}

// ---------------------------------------------------------------------------
extern "C" void kernel_launch(void* const* d_in, const int* in_sizes, int n_in,
                              void* d_out, int out_size, void* d_ws, size_t ws_size,
                              hipStream_t stream) {
    const float* embed = (const float*)d_in[0];
    const float* wih_f = (const float*)d_in[1];
    const float* whh_f = (const float*)d_in[2];
    const float* bih_f = (const float*)d_in[3];
    const float* bhh_f = (const float*)d_in[4];
    const float* wih_b = (const float*)d_in[5];
    const float* whh_b = (const float*)d_in[6];
    const float* bih_b = (const float*)d_in[7];
    const float* bhh_b = (const float*)d_in[8];
    const float* att_u = (const float*)d_in[9];
    const float* att_p = (const float*)d_in[10];
    const float* W     = (const float*)d_in[11];
    const float* bvec  = (const float*)d_in[12];
    const int*   x     = (const int*)d_in[13];
    const int*   lens  = (const int*)d_in[14];
    const int*   usr   = (const int*)d_in[15];
    const int*   prd   = (const int*)d_in[16];

    char* ws = (char*)d_ws;
    size_t off = 0;
    auto alloc = [&](size_t bytes) -> void* {
        void* p = ws + off;
        off = (off + bytes + 255) & ~(size_t)255;
        return p;
    };

    __hip_bfloat16* embed_bf = (__hip_bfloat16*)alloc((size_t)V_ * E_ * 2);
    __hip_bfloat16* wihf_bf  = (__hip_bfloat16*)alloc((size_t)G4H * E_ * 2);
    __hip_bfloat16* whhf_bf  = (__hip_bfloat16*)alloc((size_t)G4H * H_ * 2);
    __hip_bfloat16* wihb_bf  = (__hip_bfloat16*)alloc((size_t)G4H * E_ * 2);
    __hip_bfloat16* whhb_bf  = (__hip_bfloat16*)alloc((size_t)G4H * H_ * 2);
    float*          bias_f   = (float*)alloc(G4H * 4);
    float*          bias_b   = (float*)alloc(G4H * 4);
    __hip_bfloat16* Gf       = (__hip_bfloat16*)alloc((size_t)S_ * B_ * G4H * 2);
    __hip_bfloat16* Gb       = (__hip_bfloat16*)alloc((size_t)S_ * B_ * G4H * 2);
    float*          Hout     = (float*)alloc((size_t)B_ * S_ * TWO_H * 4);

    // 1) precision conversion
    cvt_f32_bf16<<<4096, 256, 0, stream>>>(embed, embed_bf, V_ * E_);
    cvt_f32_bf16<<<256, 256, 0, stream>>>(wih_f, wihf_bf, G4H * E_);
    cvt_f32_bf16<<<256, 256, 0, stream>>>(whh_f, whhf_bf, G4H * H_);
    cvt_f32_bf16<<<256, 256, 0, stream>>>(wih_b, wihb_bf, G4H * E_);
    cvt_f32_bf16<<<256, 256, 0, stream>>>(whh_b, whhb_bf, G4H * H_);
    add_f32<<<4, 256, 0, stream>>>(bih_f, bhh_f, bias_f, G4H);
    add_f32<<<4, 256, 0, stream>>>(bih_b, bhh_b, bias_b, G4H);

    // 2) hoisted input-projection GEMMs (WMMA bf16)
    input_proj<<<16384, 256, 0, stream>>>(embed_bf, wihf_bf, bias_f, x, lens, Gf, 0);
    input_proj<<<16384, 256, 0, stream>>>(embed_bf, wihb_bf, bias_b, x, lens, Gb, 1);

    // 3) persistent recurrence, one workgroup per direction (288KB LDS)
    hipFuncSetAttribute((const void*)lstm_rec,
                        hipFuncAttributeMaxDynamicSharedMemorySize, REC_SMEM);
    lstm_rec<<<2, 1024, REC_SMEM, stream>>>(whhf_bf, whhb_bf, Gf, Gb, lens, Hout);

    // 4) attention + output projection
    attn_out<<<B_, 512, 0, stream>>>(Hout, att_u, att_p, W, bvec,
                                     lens, usr, prd, (float*)d_out);
}
